// Attention_49838800503129
// MI455X (gfx1250) — compile-verified, matrix operand order checked by reading, not run on hardware
//
#include <hip/hip_runtime.h>
#include <hip/hip_bf16.h>

// ---------------------------------------------------------------------------
// MHA forward for gfx1250 (MI455X).
//  - all matmuls via v_wmma_f32_16x16x32_f16 (f32 accum)
//  - tile staging via TDM (tensor_load_to_lds, TENSORcnt) with D#-padded
//    LDS rows (stride 72 halves -> conflict-free b128 fragment loads)
//  - flash-attention streaming softmax (no [B,H,S,S] materialization)
// ---------------------------------------------------------------------------

typedef _Float16 half_t;
typedef __attribute__((ext_vector_type(16))) _Float16 v16h;
typedef __attribute__((ext_vector_type(8)))  _Float16 v8h;
typedef __attribute__((ext_vector_type(8)))  float    v8f;
typedef __attribute__((ext_vector_type(4)))  unsigned v4u;
typedef __attribute__((ext_vector_type(8)))  unsigned v8u;

#define B_   2
#define S_   2048
#define DIM_ 1024
#define H_   16
#define DH_  64
#define TOK_ (B_ * S_)           // 4096 rows

#define LDH 72                   // padded LDS row stride in halves

__device__ __forceinline__ v8f wmma_f16(v16h a, v16h b, v8f c) {
  // 8 args: (neg_a, A, neg_b, B, c_mod, C, reuse_a, reuse_b)
  return __builtin_amdgcn_wmma_f32_16x16x32_f16(false, a, false, b, (short)0, c,
                                                false, false);
}

// Load a 16x32 f16 WMMA fragment from a row-major LDS tile (stride `ld`).
// Wave32 layout (ISA 7.12.2): lane L owns row (L&15); elements 0..7 are
// K = 8*(L>>4)+0..7, elements 8..15 are K = 16+8*(L>>4)+0..7.
__device__ __forceinline__ v16h frag_ld(const half_t* p, int ld) {
  int lane = threadIdx.x & 31;
  const half_t* row = p + (lane & 15) * ld + ((lane >> 4) << 3);
  v8h lo = *(const v8h*)(row);
  v8h hi = *(const v8h*)(row + 16);
  v16h out;
#pragma unroll
  for (int i = 0; i < 8; ++i) { out[i] = lo[i]; out[i + 8] = hi[i]; }
  return out;
}

// ---------------------------------------------------------------------------
// TDM: async 2-D tile (f16 elements) global -> LDS, padded rows.
// D# group0/group1 per CDNA5 ISA.
// Rows of tile_d0 elements, tile_d1 rows, row stride `stride_elts` in memory;
// LDS gets pad_amount(+1) DWORDs appended every 2^(pad_interval+1) DWORDs.
// Issue from ONE wave only; completion via s_wait_tensorcnt.
// ---------------------------------------------------------------------------
__device__ __forceinline__ void tdm_load_2d(unsigned lds_off, const void* gptr,
                                            unsigned tile_d0, unsigned tile_d1,
                                            unsigned stride_elts,
                                            unsigned pad_interval_code,
                                            unsigned pad_amount_code) {
  unsigned long long g = (unsigned long long)(uintptr_t)gptr;
  v4u g0;
  g0[0] = 1u;                                   // count=1 (valid), user mode
  g0[1] = lds_off;                              // LDS byte address
  g0[2] = (unsigned)g;                          // global addr [31:0]
  g0[3] = (unsigned)(g >> 32) | (2u << 30);     // addr [56:32] | type=2
  v8u g1;
  g1[0] = (1u << 16)                            // data_size = 2 bytes
        | (1u << 20)                            // pad_enable
        | (pad_interval_code << 22)             // DWORDs between pads
        | (pad_amount_code << 25);              // pad DWORDs
  g1[1] = (tile_d0 & 0xFFFFu) << 16;            // tensor_dim0 = tile_d0 (lo16)
  g1[2] = (tile_d1 & 0xFFFFu) << 16;            // dim0 hi=0 | tensor_dim1 lo16
  g1[3] = (tile_d0 & 0xFFFFu) << 16;            // dim1 hi=0 | tile_dim0
  g1[4] = (tile_d1 & 0xFFFFu);                  // tile_dim1 | tile_dim2=0
  g1[5] = stride_elts;                          // tensor_dim0_stride lo32
  g1[6] = 0u;                                   // stride hi | dim1_stride lo
  g1[7] = 0u;
  asm volatile("tensor_load_to_lds %0, %1" ::"s"(g0), "s"(g1) : "memory");
}

__device__ __forceinline__ unsigned lds_off_of(const void* p) {
  // LDS aperture: flat addr[31:0] is the LDS byte offset
  return (unsigned)(uintptr_t)p;
}

// ------------------------------ cast kernels -------------------------------

__global__ __launch_bounds__(256) void cast_f32_to_f16_k(
    const float* __restrict__ in, half_t* __restrict__ out, int n4) {
  int i = blockIdx.x * blockDim.x + threadIdx.x;
  if (i < n4) {
    float4 v = ((const float4*)in)[i];
    half_t* o = out + 4 * i;
    o[0] = (half_t)v.x; o[1] = (half_t)v.y; o[2] = (half_t)v.z; o[3] = (half_t)v.w;
  }
}

// Wt[n*K + k] = (f16) W[k*N + n]   (tiled, coalesced both ways)
__global__ __launch_bounds__(256) void transpose_cast_k(
    const float* __restrict__ W, half_t* __restrict__ Wt, int K, int N) {
  __shared__ half_t tile[32][33];
  int bx = blockIdx.x * 32;              // n
  int by = blockIdx.y * 32;              // k
  int tx = threadIdx.x & 31, ty = threadIdx.x >> 5;   // ty: 0..7
#pragma unroll
  for (int i = 0; i < 32; i += 8)
    tile[ty + i][tx] = (half_t)W[(size_t)(by + ty + i) * N + bx + tx];
  __syncthreads();
#pragma unroll
  for (int i = 0; i < 32; i += 8)
    Wt[(size_t)(bx + ty + i) * K + by + tx] = tile[tx][ty + i];
}

// ------------------------------ GEMM kernel --------------------------------
// C[M,N] = A[M,K] (f16,row-major) x Bt[N,K] (f16, pre-transposed) + bias[N]
// Block tile 128x64, BK=64, 8 waves (4x2), each wave 32x32 = 2x2 wmma accums,
// 8 wmma per K-step. A/B tiles double-buffered via TDM (wave 0 issues DMA).
template <bool OUT_F32>
__global__ __launch_bounds__(256) void gemm_bias_k(
    const half_t* __restrict__ A, const half_t* __restrict__ Bt,
    const float* __restrict__ bias, void* __restrict__ C,
    int M, int N, int K) {
  __shared__ alignas(128) half_t sA[2][128][LDH];
  __shared__ alignas(128) half_t sB[2][64][LDH];

  int tid  = threadIdx.x;
  int lane = tid & 31;
  int wave = tid >> 5;
  int mw = wave >> 1;              // 0..3
  int nw = wave & 1;               // 0..1
  int m0 = blockIdx.y * 128;
  int n0 = blockIdx.x * 64;
  const int NB = K / 64;

  v8f vzero = {0.f, 0.f, 0.f, 0.f, 0.f, 0.f, 0.f, 0.f};
  v8f acc[2][2];
#pragma unroll
  for (int i = 0; i < 2; ++i)
#pragma unroll
    for (int j = 0; j < 2; ++j) acc[i][j] = vzero;

  // rows are 64 halves = 32 DWORDs (pad_interval code 4); pad 4 DWORDs
  // = 8 halves (pad_amount code 3) -> LDS stride 72 halves.
  if (wave == 0) {
    tdm_load_2d(lds_off_of(&sA[0][0][0]), A + (size_t)m0 * K, 64, 128, K, 4, 3);
    tdm_load_2d(lds_off_of(&sB[0][0][0]), Bt + (size_t)n0 * K, 64, 64, K, 4, 3);
  }

  for (int kb = 0; kb < NB; ++kb) {
    __syncthreads();   // prior reads of buffer (kb+1)&1 complete
    if (wave == 0) {
      if (kb + 1 < NB) {
        int k1 = (kb + 1) * 64;
        tdm_load_2d(lds_off_of(&sA[(kb + 1) & 1][0][0]),
                    A + (size_t)m0 * K + k1, 64, 128, K, 4, 3);
        tdm_load_2d(lds_off_of(&sB[(kb + 1) & 1][0][0]),
                    Bt + (size_t)n0 * K + k1, 64, 64, K, 4, 3);
        __builtin_amdgcn_s_wait_tensorcnt(2);   // tile kb complete (in-order)
      } else {
        __builtin_amdgcn_s_wait_tensorcnt(0);
      }
    }
    __syncthreads();

    int bf = kb & 1;
#pragma unroll
    for (int mi = 0; mi < 2; ++mi) {
      v16h a0 = frag_ld(&sA[bf][mw * 32 + mi * 16][0], LDH);
      v16h a1 = frag_ld(&sA[bf][mw * 32 + mi * 16][32], LDH);
#pragma unroll
      for (int ni = 0; ni < 2; ++ni) {
        v16h b0 = frag_ld(&sB[bf][nw * 32 + ni * 16][0], LDH);
        v16h b1 = frag_ld(&sB[bf][nw * 32 + ni * 16][32], LDH);
        acc[mi][ni] = wmma_f16(a0, b0, acc[mi][ni]);
        acc[mi][ni] = wmma_f16(a1, b1, acc[mi][ni]);
      }
    }
  }

  // epilogue: C layout -> lane (L>>4) selects row half, (L&15) is column
  int rsel = lane >> 4, csel = lane & 15;
#pragma unroll
  for (int mi = 0; mi < 2; ++mi)
#pragma unroll
    for (int ni = 0; ni < 2; ++ni) {
      int col = n0 + nw * 32 + ni * 16 + csel;
      float bv = bias[col];
#pragma unroll
      for (int j = 0; j < 8; ++j) {
        int row = m0 + mw * 32 + mi * 16 + j + 8 * rsel;
        float v = acc[mi][ni][j] + bv;
        if (OUT_F32)
          ((float*)C)[(size_t)row * N + col] = v;
        else
          ((half_t*)C)[(size_t)row * N + col] = (half_t)v;
      }
    }
}

// --------------------------- flash attention -------------------------------
// grid.x = B*H, grid.y = S/64. 128 threads = 4 waves, wave w owns q-rows
// [64*blockIdx.y + 16w, +16). Q + double-buffered K tiles staged by TDM;
// V staged manually (needs transpose for the P@V B-fragments).
__global__ __launch_bounds__(128) void flash_attn_k(
    const half_t* __restrict__ Qh, const half_t* __restrict__ Kh,
    const half_t* __restrict__ Vh, const float* __restrict__ mask,
    half_t* __restrict__ ctx) {
  const int HD = H_ * DH_;                       // 1024
  int bh = blockIdx.x;
  int b = bh >> 4, h = bh & 15;
  int q0 = blockIdx.y * 64;
  const int NB = S_ / 64;

  __shared__ alignas(128) half_t sQ[64][LDH];
  __shared__ alignas(128) half_t sK[2][64][LDH];
  __shared__ alignas(128) half_t sVT[64][LDH];   // [d][key]
  __shared__ alignas(128) half_t sP[4][16][LDH]; // per-wave P scratch
  __shared__ float sM[64];

  int tid = threadIdx.x, lane = tid & 31, wave = tid >> 5;
  int rsel = lane >> 4, csel = lane & 15;
  size_t base = ((size_t)b * S_) * HD + (size_t)h * DH_;

  if (wave == 0) {
    tdm_load_2d(lds_off_of(&sQ[0][0]), Qh + base + (size_t)q0 * HD,
                64, 64, HD, 4, 3);
    tdm_load_2d(lds_off_of(&sK[0][0][0]), Kh + base, 64, 64, HD, 4, 3);
    __builtin_amdgcn_s_wait_tensorcnt(1);        // Q complete, K0 in flight
  }
  __syncthreads();
  v16h aq0 = frag_ld(&sQ[wave * 16][0], LDH);    // d = 0..31
  v16h aq1 = frag_ld(&sQ[wave * 16][32], LDH);   // d = 32..63

  v8f vzero = {0.f, 0.f, 0.f, 0.f, 0.f, 0.f, 0.f, 0.f};
  v8f o[4];
#pragma unroll
  for (int t = 0; t < 4; ++t) o[t] = vzero;
  float rm[8], rl[8];
#pragma unroll
  for (int j = 0; j < 8; ++j) { rm[j] = -1e30f; rl[j] = 0.f; }
  const float scale = 0.125f;                    // 1/sqrt(64)

  for (int kb = 0; kb < NB; ++kb) {
    int k0 = kb * 64;
    __syncthreads();
    // issue next K tile DMA, overlap with manual V-transpose staging
    if (wave == 0 && kb + 1 < NB)
      tdm_load_2d(lds_off_of(&sK[(kb + 1) & 1][0][0]),
                  Kh + base + (size_t)(k0 + 64) * HD, 64, 64, HD, 4, 3);
    for (int i = tid; i < 512; i += 128) {
      int r = i >> 3, c = (i & 7) * 8;
      v8h v = *(const v8h*)(Vh + base + (size_t)(k0 + r) * HD + c);
#pragma unroll
      for (int e = 0; e < 8; ++e) sVT[c + e][r] = v[e];
      if (k0 + 64 < S_)
        __builtin_prefetch(Vh + base + (size_t)(k0 + 64 + r) * HD + c, 0, 0);
    }
    if (tid < 64) sM[tid] = mask[(size_t)b * S_ + k0 + tid];
    if (wave == 0) {
      if (kb + 1 < NB) __builtin_amdgcn_s_wait_tensorcnt(1);  // K(kb) done
      else             __builtin_amdgcn_s_wait_tensorcnt(0);
    }
    __syncthreads();

    // scores: S = Q x K^T  (B-frag lane = key, elements walk d)
    const half_t* kbuf = &sK[kb & 1][0][0];
    v8f s[4];
#pragma unroll
    for (int t = 0; t < 4; ++t) {
      v16h bk0 = frag_ld(kbuf + (t * 16) * LDH, LDH);
      v16h bk1 = frag_ld(kbuf + (t * 16) * LDH + 32, LDH);
      v8f c = wmma_f16(aq0, bk0, vzero);
      s[t] = wmma_f16(aq1, bk1, c);
    }

    float pen[4];
#pragma unroll
    for (int t = 0; t < 4; ++t)
      pen[t] = -1000000.0f * (1.0f - sM[t * 16 + csel]);

    // online softmax over this key tile
#pragma unroll
    for (int j = 0; j < 8; ++j) {
      float x0 = s[0][j] * scale + pen[0];
      float x1 = s[1][j] * scale + pen[1];
      float x2 = s[2][j] * scale + pen[2];
      float x3 = s[3][j] * scale + pen[3];
      float mx = fmaxf(fmaxf(x0, x1), fmaxf(x2, x3));
#pragma unroll
      for (int off = 1; off < 16; off <<= 1)
        mx = fmaxf(mx, __shfl_xor(mx, off, 32));
      float mnew  = fmaxf(rm[j], mx);
      float alpha = __expf(rm[j] - mnew);
      float p0 = __expf(x0 - mnew), p1 = __expf(x1 - mnew);
      float p2 = __expf(x2 - mnew), p3 = __expf(x3 - mnew);
      float rs = (p0 + p1) + (p2 + p3);
#pragma unroll
      for (int off = 1; off < 16; off <<= 1)
        rs += __shfl_xor(rs, off, 32);
      rm[j] = mnew;
      rl[j] = rl[j] * alpha + rs;
#pragma unroll
      for (int t = 0; t < 4; ++t) o[t][j] *= alpha;
      s[0][j] = p0; s[1][j] = p1; s[2][j] = p2; s[3][j] = p3;
    }

    // C-layout -> A-layout for P via per-wave LDS scratch
#pragma unroll
    for (int t = 0; t < 4; ++t)
#pragma unroll
      for (int j = 0; j < 8; ++j)
        sP[wave][j + 8 * rsel][t * 16 + csel] = (half_t)s[t][j];
    __syncthreads();

    // O += P x V  (B-frag lane = d, elements walk keys; sVT is [d][key])
    v16h ap0 = frag_ld(&sP[wave][0][0], LDH);    // keys 0..31
    v16h ap1 = frag_ld(&sP[wave][0][32], LDH);   // keys 32..63
#pragma unroll
    for (int t = 0; t < 4; ++t) {
      v16h bv0 = frag_ld(&sVT[t * 16][0], LDH);
      v16h bv1 = frag_ld(&sVT[t * 16][32], LDH);
      o[t] = wmma_f16(ap0, bv0, o[t]);
      o[t] = wmma_f16(ap1, bv1, o[t]);
    }
  }

  // normalize + write ctx (f16, combined-heads layout [tok, h*DH+d])
#pragma unroll
  for (int t = 0; t < 4; ++t)
#pragma unroll
    for (int j = 0; j < 8; ++j) {
      int row = q0 + wave * 16 + j + 8 * rsel;
      float v = o[t][j] / rl[j];
      ctx[base + (size_t)row * HD + t * 16 + csel] = (half_t)v;
    }
}

// ------------------------------- launcher ----------------------------------

extern "C" void kernel_launch(void* const* d_in, const int* in_sizes, int n_in,
                              void* d_out, int out_size, void* d_ws,
                              size_t ws_size, hipStream_t stream) {
  const float* X    = (const float*)d_in[0];
  const float* mask = (const float*)d_in[1];
  const float* Wq   = (const float*)d_in[2];
  const float* bq   = (const float*)d_in[3];
  const float* Wk   = (const float*)d_in[4];
  const float* bk   = (const float*)d_in[5];
  const float* Wv   = (const float*)d_in[6];
  const float* bv   = (const float*)d_in[7];
  const float* Wo   = (const float*)d_in[8];
  const float* bo   = (const float*)d_in[9];
  float* out = (float*)d_out;

  const size_t MSZ = (size_t)TOK_ * DIM_;        // 4096*1024 elements
  const size_t WSZ = (size_t)DIM_ * DIM_;        // 1024*1024 elements
  half_t* Xh  = (half_t*)d_ws;
  half_t* Wqt = Xh + MSZ;
  half_t* Wkt = Wqt + WSZ;
  half_t* Wvt = Wkt + WSZ;
  half_t* Wot = Wvt + WSZ;
  half_t* Qh  = Wot + WSZ;
  half_t* Kh  = Qh + MSZ;
  half_t* Vh  = Kh + MSZ;
  half_t* Ch  = Vh + MSZ;

  // 1) cast X to f16
  cast_f32_to_f16_k<<<(int)(MSZ / 4 / 256), 256, 0, stream>>>(X, Xh, (int)(MSZ / 4));

  // 2) transpose-cast weights to [N][K] f16
  dim3 tg(DIM_ / 32, DIM_ / 32);
  transpose_cast_k<<<tg, 256, 0, stream>>>(Wq, Wqt, DIM_, DIM_);
  transpose_cast_k<<<tg, 256, 0, stream>>>(Wk, Wkt, DIM_, DIM_);
  transpose_cast_k<<<tg, 256, 0, stream>>>(Wv, Wvt, DIM_, DIM_);
  transpose_cast_k<<<tg, 256, 0, stream>>>(Wo, Wot, DIM_, DIM_);

  // 3) QKV projections (f16 out)
  dim3 gg(DIM_ / 64, TOK_ / 128);
  gemm_bias_k<false><<<gg, 256, 0, stream>>>(Xh, Wqt, bq, Qh, TOK_, DIM_, DIM_);
  gemm_bias_k<false><<<gg, 256, 0, stream>>>(Xh, Wkt, bk, Kh, TOK_, DIM_, DIM_);
  gemm_bias_k<false><<<gg, 256, 0, stream>>>(Xh, Wvt, bv, Vh, TOK_, DIM_, DIM_);

  // 4) flash attention -> ctx (f16)
  dim3 fa(B_ * H_, S_ / 64);
  flash_attn_k<<<fa, 128, 0, stream>>>(Qh, Kh, Vh, mask, Ch);

  // 5) output projection (f32 out)
  gemm_bias_k<true><<<gg, 256, 0, stream>>>(Ch, Wot, bo, out, TOK_, DIM_, DIM_);
}
